// CrissCrossAttention_27530740367925
// MI455X (gfx1250) — matrix operand order repaired and checked by reading, not compiled, not run on hardware
//
#include <hip/hip_runtime.h>
#include <math.h>

// ---------------------------------------------------------------------------
// CrissCrossAttention for MI455X (gfx1250, wave32, WMMA + TDM).
// All GEMMs on v_wmma_f32_16x16x32_bf16; layout-preserving LDS stages use the
// Tensor Data Mover (tensor_load_to_lds + s_wait_tensorcnt). 128 MB workspace
// stays inside the 192 MB L2.
// ---------------------------------------------------------------------------

typedef __bf16 bf16;
typedef __attribute__((ext_vector_type(8)))  bf16  v8bf;
typedef __attribute__((ext_vector_type(16))) bf16  v16bf;
typedef __attribute__((ext_vector_type(8)))  float v8f;
typedef __attribute__((ext_vector_type(4)))  unsigned int v4u;
typedef __attribute__((ext_vector_type(4)))  int   v4i;
typedef __attribute__((ext_vector_type(8)))  int   v8i;

constexpr int B_  = 8;
constexpr int C_  = 512;
constexpr int C8_ = 64;
constexpr int H_  = 64;
constexpr int W_  = 64;
constexpr int HW_ = H_ * W_;

#define DEVFN static __device__ __forceinline__

#if __has_builtin(__builtin_amdgcn_tensor_load_to_lds) && \
    __has_builtin(__builtin_amdgcn_s_wait_tensorcnt)
#define HAVE_TDM 1
#else
#define HAVE_TDM 0
#endif

DEVFN bf16 tobf(float f) { return (bf16)f; }

DEVFN v16bf join16(const bf16* p0, const bf16* p1) {
  v8bf lo = *(const v8bf*)p0;   // ds_load_b128
  v8bf hi = *(const v8bf*)p1;   // ds_load_b128
  return __builtin_shufflevector(lo, hi, 0,1,2,3,4,5,6,7,8,9,10,11,12,13,14,15);
}

// A fragment (16x32 bf16), LDS row-major [m][k], row stride ld elements.
// lanes 0-15: M=lane, K 0-7 & 16-23 ; lanes 16-31: M=lane-16, K 8-15 & 24-31.
DEVFN v16bf load_a(const bf16* lds, int ld, int lane) {
  const int m  = lane & 15;
  const int kb = (lane < 16) ? 0 : 8;
  const bf16* r = lds + m * ld + kb;
  return join16(r, r + 16);
}

// B fragment (32x16 bf16), LDS stored transposed [n][k], row stride ld.
// lanes 0-15: N=lane, K 0-7 & 16-23 ; lanes 16-31: N=lane-16, K 8-15 & 24-31.
DEVFN v16bf load_b(const bf16* lds, int ld, int lane) {
  const int n  = lane & 15;
  const int kb = (lane < 16) ? 0 : 8;
  const bf16* r = lds + n * ld + kb;
  return join16(r, r + 16);
}

DEVFN v8f wmma_bf16(v16bf a, v16bf b, v8f c) {
  return __builtin_amdgcn_wmma_f32_16x16x32_bf16(false, a, false, b, (short)0, c,
                                                 false, false);
}

#if HAVE_TDM
// TDM: async-copy a 2D tile (tile_d0 contiguous bf16 elems x tile_d1 rows,
// row stride stride_d0 elems) from global into LDS (packed rows). ISA §8.3-8.4.
DEVFN void tdm_load_2d_bf16(const void* lds_ptr, const void* gptr,
                            unsigned tile_d0, unsigned tile_d1,
                            unsigned long long stride_d0) {
  unsigned long long ga = (unsigned long long)gptr;
  v4u g0;
  g0[0] = 1u;                                            // count=1, user mode
  g0[1] = (unsigned)(unsigned long long)lds_ptr;         // LDS byte address
  g0[2] = (unsigned)(ga & 0xFFFFFFFFu);                  // global_addr[31:0]
  g0[3] = (unsigned)((ga >> 32) & 0x01FFFFFFu) | (2u << 30);  // [56:32], type=2
  unsigned long long td0 = tile_d0, td1 = tile_d1;       // tensor dims = tile dims
  v8i g1;
  g1[0] = (int)(1u << 16);                               // data_size=1 (2 bytes)
  g1[1] = (int)((td0 & 0xFFFFu) << 16);                  // tensor_dim0[15:0]
  g1[2] = (int)(((td0 >> 16) & 0xFFFFu) | ((td1 & 0xFFFFu) << 16));
  g1[3] = (int)(((td1 >> 16) & 0xFFFFu) | ((tile_d0 & 0xFFFFu) << 16));
  g1[4] = (int)(tile_d1 & 0xFFFFu);                      // tile_dim1; tile_dim2=0
  g1[5] = (int)(stride_d0 & 0xFFFFFFFFu);                // dim0_stride[31:0]
  g1[6] = (int)((stride_d0 >> 32) & 0xFFFFu);            // dim0_stride[47:32]
  g1[7] = 0;
  v4i z4 = {0, 0, 0, 0};
#if defined(__clang_major__) && (__clang_major__ >= 23)
  v8i z8 = {0, 0, 0, 0, 0, 0, 0, 0};
  __builtin_amdgcn_tensor_load_to_lds(g0, g1, z4, z4, z8, 0);
#else
  __builtin_amdgcn_tensor_load_to_lds(g0, g1, z4, z4, 0);
#endif
}
#endif

// ---------------------------------------------------------------------------
// 1) Projection GEMM: out_bf16[b][o][n] = sum_c Wm[o][c]*in[b][c][n] + bias[o]
//    Block tile 64(M) x 64(N), 4 waves; each wave: 1 A-frag x 4 B-frags ->
//    4 WMMA per K-step. grid = (HW/64, O/64, B), block = 128.
// ---------------------------------------------------------------------------
template <int O>
__global__ __launch_bounds__(128) void proj_kernel(const float* __restrict__ in,
                                                   const float* __restrict__ Wm,
                                                   const float* __restrict__ bias,
                                                   bf16* __restrict__ out) {
  __shared__ alignas(16) bf16 As[64 * 32];   // [m][k]
  __shared__ alignas(16) bf16 Bs[64 * 32];   // [n][k]  (transposed stage)
  const int t = threadIdx.x, wave = t >> 5, lane = t & 31;
  const int nblk = blockIdx.x * 64;
  const int o0   = blockIdx.y * 64;
  const int b    = blockIdx.z;
  const float* inb = in + (size_t)b * C_ * HW_;
  const int m0 = wave * 16;

  v8f acc[4] = {};
  for (int k0 = 0; k0 < C_; k0 += 32) {
#pragma unroll
    for (int r = 0; r < 16; ++r) {           // A: 2048 elems, coalesced rows
      int idx = r * 128 + t;
      int m = idx >> 5, kk = idx & 31;
      As[m * 32 + kk] = tobf(Wm[(size_t)(o0 + m) * C_ + k0 + kk]);
    }
#pragma unroll
    for (int r = 0; r < 16; ++r) {           // B: 2048 elems, coalesced n
      int idx = r * 128 + t;
      int kk = idx >> 6, n = idx & 63;
      Bs[n * 32 + kk] = tobf(inb[(size_t)(k0 + kk) * HW_ + nblk + n]);
    }
    __syncthreads();
    v16bf a = load_a(As + m0 * 32, 32, lane);
#pragma unroll
    for (int j = 0; j < 4; ++j) {
      v16bf bb = load_b(Bs + (j * 16) * 32, 32, lane);
      acc[j] = wmma_bf16(a, bb, acc[j]);
    }
    __syncthreads();
    if (k0 + 32 < C_)                        // -> global_prefetch_b8
      __builtin_prefetch(&inb[(size_t)(k0 + 32 + (t >> 6)) * HW_ + nblk + (t & 63)], 0, 1);
  }
  const int mh = (lane < 16) ? 0 : 8;
#pragma unroll
  for (int j = 0; j < 4; ++j) {
    const int n = nblk + j * 16 + (lane & 15);
#pragma unroll
    for (int r = 0; r < 8; ++r) {
      int o = o0 + m0 + r + mh;
      out[((size_t)b * O + o) * HW_ + n] = tobf(acc[j][r] + bias[o]);
    }
  }
}

// ---------------------------------------------------------------------------
// 2) eH[b,h,w,k] = sum_c pq[b,c,h,w] * pk[b,c,k,w]   (fixed b,w; 64x64x64)
// ---------------------------------------------------------------------------
__global__ __launch_bounds__(128) void eH_kernel(const bf16* __restrict__ pq,
                                                 const bf16* __restrict__ pk,
                                                 float* __restrict__ e) {
  __shared__ alignas(16) bf16 Aq[64 * 64];   // [h][c]
  __shared__ alignas(16) bf16 Bk[64 * 64];   // [kh][c]
  const int t = threadIdx.x, wave = t >> 5, lane = t & 31;
  const int w = blockIdx.x, b = blockIdx.y;
#pragma unroll
  for (int r = 0; r < 32; ++r) {
    int idx = r * 128 + t;
    int c = idx >> 6, h = idx & 63;
    size_t g = (((size_t)b * C8_ + c) * H_ + h) * W_ + w;
    Aq[h * 64 + c] = pq[g];
    Bk[h * 64 + c] = pk[g];
  }
  __syncthreads();
  const int m0 = wave * 16;
  v8f acc[4] = {};
#pragma unroll
  for (int k0 = 0; k0 < 64; k0 += 32) {
    v16bf a = load_a(Aq + m0 * 64 + k0, 64, lane);
#pragma unroll
    for (int j = 0; j < 4; ++j) {
      v16bf bb = load_b(Bk + (j * 16) * 64 + k0, 64, lane);
      acc[j] = wmma_bf16(a, bb, acc[j]);
    }
  }
  const int mh = (lane < 16) ? 0 : 8, n0 = lane & 15;
#pragma unroll
  for (int j = 0; j < 4; ++j)
#pragma unroll
    for (int r = 0; r < 8; ++r) {
      int h = m0 + r + mh, k = j * 16 + n0;
      e[(((size_t)b * H_ + h) * W_ + w) * 128 + k] = acc[j][r];
    }
}

// ---------------------------------------------------------------------------
// 3) eW[b,h,w,k] = sum_c pq[b,c,h,w] * pk[b,c,h,k]   (fixed b,h)
// ---------------------------------------------------------------------------
__global__ __launch_bounds__(128) void eW_kernel(const bf16* __restrict__ pq,
                                                 const bf16* __restrict__ pk,
                                                 float* __restrict__ e) {
  __shared__ alignas(16) bf16 Aq[64 * 64];   // [wq][c]
  __shared__ alignas(16) bf16 Bk[64 * 64];   // [wk][c]
  const int t = threadIdx.x, wave = t >> 5, lane = t & 31;
  const int h = blockIdx.x, b = blockIdx.y;
#pragma unroll
  for (int r = 0; r < 32; ++r) {
    int idx = r * 128 + t;
    int c = idx >> 6, wq = idx & 63;
    size_t g = (((size_t)b * C8_ + c) * H_ + h) * W_ + wq;
    Aq[wq * 64 + c] = pq[g];
    Bk[wq * 64 + c] = pk[g];
  }
  __syncthreads();
  const int m0 = wave * 16;
  v8f acc[4] = {};
#pragma unroll
  for (int k0 = 0; k0 < 64; k0 += 32) {
    v16bf a = load_a(Aq + m0 * 64 + k0, 64, lane);
#pragma unroll
    for (int j = 0; j < 4; ++j) {
      v16bf bb = load_b(Bk + (j * 16) * 64 + k0, 64, lane);
      acc[j] = wmma_bf16(a, bb, acc[j]);
    }
  }
  const int mh = (lane < 16) ? 0 : 8, n0 = lane & 15;
#pragma unroll
  for (int j = 0; j < 4; ++j)
#pragma unroll
    for (int r = 0; r < 8; ++r) {
      int wq = m0 + r + mh, k = j * 16 + n0;
      e[(((size_t)b * H_ + h) * W_ + wq) * 128 + 64 + k] = acc[j][r];
    }
}

// ---------------------------------------------------------------------------
// 4) softmax over 128 logits per (b,h,w); diag mask eH[k==h] = -inf
// ---------------------------------------------------------------------------
__global__ __launch_bounds__(256) void softmax_kernel(const float* __restrict__ e,
                                                      bf16* __restrict__ att) {
  const int idx = blockIdx.x * 256 + threadIdx.x;
  const int h = (idx >> 6) & 63;
  const float* row = e + (size_t)idx * 128;
  float x[128];
#pragma unroll
  for (int i = 0; i < 128; i += 4) {
    float4 v4 = *(const float4*)(row + i);
    x[i + 0] = v4.x; x[i + 1] = v4.y; x[i + 2] = v4.z; x[i + 3] = v4.w;
  }
#pragma unroll
  for (int i = 0; i < 64; ++i)
    x[i] = (i == h) ? -__builtin_inff() : x[i];
  float m = -__builtin_inff();
#pragma unroll
  for (int i = 0; i < 128; ++i) m = fmaxf(m, x[i]);
  float s = 0.f;
#pragma unroll
  for (int i = 0; i < 128; ++i) { float p = __expf(x[i] - m); x[i] = p; s += p; }
  const float inv = 1.0f / s;
  bf16* dst = att + (size_t)idx * 128;
#pragma unroll
  for (int i = 0; i < 128; i += 8) {
    v8bf ob;
#pragma unroll
    for (int j = 0; j < 8; ++j) ob[j] = tobf(x[i + j] * inv);
    *(v8bf*)(dst + i) = ob;
  }
}

// ---------------------------------------------------------------------------
// 5) outH[b,c,h,w] = sum_k pv[b,c,k,w] * aH[b,h,w,k]  (fixed b,w)
//    Bh tile is layout-preserving -> TDM; Av is a column gather -> manual.
// ---------------------------------------------------------------------------
__global__ __launch_bounds__(128) void outH_kernel(const bf16* __restrict__ pv,
                                                   const bf16* __restrict__ att,
                                                   float* __restrict__ oH) {
  __shared__ alignas(16) bf16 Av[64 * 64];   // [cl][kh]
  __shared__ alignas(16) bf16 Bh[64 * 64];   // [h][kh]
  const int t = threadIdx.x, wave = t >> 5, lane = t & 31;
  const int c0 = blockIdx.x * 64, w = blockIdx.y, b = blockIdx.z;

#if HAVE_TDM
  if (wave == 0) {                          // one TDM issue per workgroup
    tdm_load_2d_bf16(Bh, &att[(((size_t)b * H_) * W_ + w) * 128],
                     /*tile_d0=*/64, /*tile_d1=*/64, /*stride=*/(size_t)W_ * 128);
  }
#endif
#pragma unroll
  for (int r = 0; r < 32; ++r) {
    int idx = r * 128 + t;
    int i0 = idx >> 6, i1 = idx & 63;
    Av[i0 * 64 + i1] = pv[(((size_t)b * C_ + c0 + i0) * H_ + i1) * W_ + w];
#if !HAVE_TDM
    Bh[i0 * 64 + i1] = att[(((size_t)b * H_ + i0) * W_ + w) * 128 + i1];
#endif
  }
#if HAVE_TDM
  if (wave == 0) __builtin_amdgcn_s_wait_tensorcnt(0);
#endif
  __syncthreads();
  const int m0 = wave * 16;
  v8f acc[4] = {};
#pragma unroll
  for (int k0 = 0; k0 < 64; k0 += 32) {
    v16bf a = load_a(Av + m0 * 64 + k0, 64, lane);
#pragma unroll
    for (int j = 0; j < 4; ++j) {
      v16bf bb = load_b(Bh + (j * 16) * 64 + k0, 64, lane);
      acc[j] = wmma_bf16(a, bb, acc[j]);
    }
  }
  const int mh = (lane < 16) ? 0 : 8, n0 = lane & 15;
#pragma unroll
  for (int j = 0; j < 4; ++j)
#pragma unroll
    for (int r = 0; r < 8; ++r) {
      int c = c0 + m0 + r + mh, hh = j * 16 + n0;
      oH[(((size_t)b * C_ + c) * H_ + hh) * W_ + w] = acc[j][r];
    }
}

// ---------------------------------------------------------------------------
// 6) outW + epilogue: out = gamma*(outW + oH) + v     (fixed b,h)
//    Both stages layout-preserving -> two TDM tile loads.
// ---------------------------------------------------------------------------
__global__ __launch_bounds__(128) void outW_final_kernel(const bf16* __restrict__ pv,
                                                         const bf16* __restrict__ att,
                                                         const float* __restrict__ oH,
                                                         const float* __restrict__ vin,
                                                         const float* __restrict__ gamma,
                                                         float* __restrict__ out) {
  __shared__ alignas(16) bf16 Av[64 * 64];   // [cl][kw]
  __shared__ alignas(16) bf16 Bw[64 * 64];   // [wq][kw]
  const int t = threadIdx.x, wave = t >> 5, lane = t & 31;
  const int c0 = blockIdx.x * 64, h = blockIdx.y, b = blockIdx.z;

#if HAVE_TDM
  if (wave == 0) {
    tdm_load_2d_bf16(Av, &pv[(((size_t)b * C_ + c0) * H_ + h) * W_],
                     64, 64, (size_t)H_ * W_);
    tdm_load_2d_bf16(Bw, &att[(((size_t)b * H_ + h) * W_) * 128 + 64],
                     64, 64, 128);
    __builtin_amdgcn_s_wait_tensorcnt(0);
  }
#else
#pragma unroll
  for (int r = 0; r < 32; ++r) {
    int idx = r * 128 + t;
    int i0 = idx >> 6, i1 = idx & 63;
    Av[i0 * 64 + i1] = pv[(((size_t)b * C_ + c0 + i0) * H_ + h) * W_ + i1];
    Bw[i0 * 64 + i1] = att[(((size_t)b * H_ + h) * W_ + i0) * 128 + 64 + i1];
  }
#endif
  __syncthreads();
  const int m0 = wave * 16;
  v8f acc[4] = {};
#pragma unroll
  for (int k0 = 0; k0 < 64; k0 += 32) {
    v16bf a = load_a(Av + m0 * 64 + k0, 64, lane);
#pragma unroll
    for (int j = 0; j < 4; ++j) {
      v16bf bb = load_b(Bw + (j * 16) * 64 + k0, 64, lane);
      acc[j] = wmma_bf16(a, bb, acc[j]);
    }
  }
  const float g = gamma[0];
  const int mh = (lane < 16) ? 0 : 8, n0 = lane & 15;
#pragma unroll
  for (int j = 0; j < 4; ++j)
#pragma unroll
    for (int r = 0; r < 8; ++r) {
      int c = c0 + m0 + r + mh, wq = j * 16 + n0;
      size_t idx = (((size_t)b * C_ + c) * H_ + h) * W_ + wq;
      out[idx] = g * (acc[j][r] + oH[idx]) + vin[idx];
    }
}

// ---------------------------------------------------------------------------
extern "C" void kernel_launch(void* const* d_in, const int* in_sizes, int n_in,
                              void* d_out, int out_size, void* d_ws, size_t ws_size,
                              hipStream_t stream) {
  const float* q     = (const float*)d_in[0];
  const float* k     = (const float*)d_in[1];
  const float* v     = (const float*)d_in[2];
  const float* Wq    = (const float*)d_in[3];
  const float* bq    = (const float*)d_in[4];
  const float* Wk    = (const float*)d_in[5];
  const float* bk    = (const float*)d_in[6];
  const float* Wv    = (const float*)d_in[7];
  const float* bv    = (const float*)d_in[8];
  const float* gamma = (const float*)d_in[9];
  float* out = (float*)d_out;

  // Workspace layout (128 MB total -- fits in the 192 MB L2):
  char* ws = (char*)d_ws;
  bf16*  pq  = (bf16*)(ws);                           //  4 MB  [B][64][HW]
  bf16*  pk  = (bf16*)(ws + (size_t)(4)  * 1048576);  //  4 MB
  bf16*  pv  = (bf16*)(ws + (size_t)(8)  * 1048576);  // 32 MB  [B][512][HW]
  float* e   = (float*)(ws + (size_t)(40) * 1048576); // 16 MB  [B][H][W][128]
  bf16*  att = (bf16*)(ws + (size_t)(56) * 1048576);  //  8 MB  [B][H][W][128]
  float* oH  = (float*)(ws + (size_t)(64) * 1048576); // 64 MB  [B][C][H][W]

  dim3 blk(128);
  proj_kernel<64> <<<dim3(HW_ / 64, C8_ / 64, B_), blk, 0, stream>>>(q, Wq, bq, pq);
  proj_kernel<64> <<<dim3(HW_ / 64, C8_ / 64, B_), blk, 0, stream>>>(k, Wk, bk, pk);
  proj_kernel<512><<<dim3(HW_ / 64, C_  / 64, B_), blk, 0, stream>>>(v, Wv, bv, pv);
  eH_kernel<<<dim3(W_, B_), blk, 0, stream>>>(pq, pk, e);
  eW_kernel<<<dim3(H_, B_), blk, 0, stream>>>(pq, pk, e);
  softmax_kernel<<<dim3((B_ * H_ * W_) / 256), dim3(256), 0, stream>>>(e, att);
  outH_kernel<<<dim3(C_ / 64, W_, B_), blk, 0, stream>>>(pv, att, oH);
  outW_final_kernel<<<dim3(C_ / 64, H_, B_), blk, 0, stream>>>(pv, att, oH, v, gamma, out);
}
